// AttentionBlock_76115410419715
// MI455X (gfx1250) — matrix operand order repaired and checked by reading, not compiled, not run on hardware
//
#include <hip/hip_runtime.h>

#define Bsz 4
#define Ssz 2048
#define Dsz 64
#define Hsz 12
#define DH  768           // Dsz*Hsz
#define NR  (Bsz*Ssz)     // 8192 rows

typedef __attribute__((ext_vector_type(16))) _Float16 v16h;
typedef __attribute__((ext_vector_type(8)))  _Float16 v8h;
typedef __attribute__((ext_vector_type(8)))  float    v8f;

union V16H { v16h v; int4 q[2]; _Float16 h[16]; };
union V8H  { v8h  v; int4 q; };

static __device__ __forceinline__ v8f wmma16(const V16H& a, const V16H& b, v8f c) {
  return __builtin_amdgcn_wmma_f32_16x16x32_f16(false, a.v, false, b.v, (short)0, c, false, false);
}
static __device__ __forceinline__ int4 ld16(const _Float16* p) { return *(const int4*)p; }

// ---------------- prep: fp32 -> f16 convert / transpose-convert ----------------
__global__ void cvt_half_kernel(const float* __restrict__ in, _Float16* __restrict__ out, int n) {
  int i = blockIdx.x * blockDim.x + threadIdx.x;
  if (i < n) out[i] = (_Float16)in[i];
}
// in [R][C] fp32 row-major -> out [C][R] f16 row-major
__global__ void trcvt_kernel(const float* __restrict__ in, _Float16* __restrict__ out, int R, int C) {
  int i = blockIdx.x * blockDim.x + threadIdx.x;
  if (i >= R * C) return;
  int c = i / R, r = i % R;
  out[i] = (_Float16)in[r * C + c];
}

// ---------------- kernel 1: projection  Y = (X*W + b) * scale -------------------
// x16:[NR][64] f16, wT:[768][64] f16 (W transposed). Either row-major out [NR][768]
// or transposed out [B][768][S] (for V). grid (48 n-tiles, 512 s-tiles), block 32.
__global__ void qkv_kernel(const _Float16* __restrict__ x16, const _Float16* __restrict__ wT,
                           const float* __restrict__ bias, _Float16* __restrict__ outRM,
                           _Float16* __restrict__ outTR, float scale) {
  const int lane = threadIdx.x, hi = lane >> 4, lm = lane & 15;
  const int n0 = blockIdx.x * 16, s0 = blockIdx.y * 16;
  // B tile: lane holds col n0+lm, K chunks {8*hi..}, {8*hi+16..}
  V16H b0, b1;
  const _Float16* br = wT + (size_t)(n0 + lm) * 64 + hi * 8;
  b0.q[0] = ld16(br);      b0.q[1] = ld16(br + 16);
  b1.q[0] = ld16(br + 32); b1.q[1] = ld16(br + 48);
  // A tile: lane holds row s0+lm, K chunks {8*hi..}, {8*hi+16..}
  const _Float16* ar = x16 + (size_t)(s0 + lm) * 64 + hi * 8;
  V16H a0, a1;
  a0.q[0] = ld16(ar);      a0.q[1] = ld16(ar + 16);
  a1.q[0] = ld16(ar + 32); a1.q[1] = ld16(ar + 48);
  v8f c = {};
  c = wmma16(a0, b0, c);
  c = wmma16(a1, b1, c);
  const float bn = bias[n0 + lm];
  if (outRM) {
    _Float16* o = outRM + n0 + lm;
#pragma unroll
    for (int r = 0; r < 8; r++) {
      int s = s0 + r + 8 * hi;
      o[(size_t)s * DH] = (_Float16)((c[r] + bn) * scale);
    }
  } else {  // V: store transposed [B][DH][S], 8 consecutive s pack to one b128 store
    int b = s0 >> 11, sl = (s0 & 2047) + 8 * hi;
    V8H pk;
#pragma unroll
    for (int r = 0; r < 8; r++) pk.v[r] = (_Float16)((c[r] + bn) * scale);
    *(int4*)(outTR + ((size_t)(b * DH + n0 + lm)) * Ssz + sl) = pk.q;
  }
}

// ---------------- kernel 2: inv[b,h,k] = 1 / sum_q exp(s[q,k]) ------------------
// s = qp (already scaled by 1/8) * kp^T.  grid (128 k-tiles, H, B), block 32.
__global__ void colsum_kernel(const _Float16* __restrict__ qp, const _Float16* __restrict__ kp,
                              float* __restrict__ inv) {
  const int lane = threadIdx.x, hi = lane >> 4, lm = lane & 15;
  const int k0 = blockIdx.x * 16, h = blockIdx.y, b = blockIdx.z;
  const _Float16* qb = qp + (size_t)b * Ssz * DH + h * 64;
  const _Float16* kb = kp + (size_t)b * Ssz * DH + h * 64;
  // B tiles (kp^T): lane holds col k0+lm, d chunks {d0+8*hi..}, {d0+8*hi+16..}
  V16H b0, b1;
  const _Float16* br = kb + (size_t)(k0 + lm) * DH + hi * 8;
  b0.q[0] = ld16(br);      b0.q[1] = ld16(br + 16);
  b1.q[0] = ld16(br + 32); b1.q[1] = ld16(br + 48);
  float acc = 0.f;
  for (int q0 = 0; q0 < Ssz; q0 += 16) {
    const _Float16* ar = qb + (size_t)(q0 + lm) * DH + hi * 8;
    V16H a0, a1;
    a0.q[0] = ld16(ar);      a0.q[1] = ld16(ar + 16);
    a1.q[0] = ld16(ar + 32); a1.q[1] = ld16(ar + 48);
    v8f c = {};
    c = wmma16(a0, b0, c);
    c = wmma16(a1, b1, c);
#pragma unroll
    for (int r = 0; r < 8; r++) acc += __expf(c[r]);
  }
  acc += __shfl_xor(acc, 16, 32);
  if (!hi) inv[((size_t)b * Hsz + h) * Ssz + k0 + lm] = 1.0f / acc;
}

// ---------------- kernel 2b: fold inv into V (in place, deterministic) ----------
__global__ void vscale_kernel(_Float16* __restrict__ vpT, const float* __restrict__ inv) {
  size_t t = (size_t)blockIdx.x * blockDim.x + threadIdx.x;  // one thread per 8 s-values
  size_t o = t * 8;
  int b = (int)(o / ((size_t)DH * Ssz));
  size_t rem = o % ((size_t)DH * Ssz);
  int hd = (int)(rem / Ssz), s = (int)(rem % Ssz);
  const float* ip = inv + ((size_t)b * Hsz + (hd >> 6)) * Ssz + s;
  V8H x; x.q = *(int4*)(vpT + o);
#pragma unroll
  for (int i = 0; i < 8; i++) x.v[i] = (_Float16)((float)x.v[i] * ip[i]);
  *(int4*)(vpT + o) = x.q;
}

// ---------------- kernel 3: ctx^T[d,q] = sum_k v'[k,d] * exp(s'[k,q]) -----------
// s' computed transposed (A=kp rows, B=qp^T) so the two stacked 16x16 C-tiles
// repack into the 32x16 f16 B operand with only shfl_xor(16). grid (128,H,B).
__global__ void ctx_kernel(const _Float16* __restrict__ qp, const _Float16* __restrict__ kp,
                           const _Float16* __restrict__ vpT, _Float16* __restrict__ ctx) {
  const int lane = threadIdx.x, hi = lane >> 4, lm = lane & 15;
  const int q0 = blockIdx.x * 16, h = blockIdx.y, b = blockIdx.z;
  const _Float16* qb = qp + (size_t)b * Ssz * DH + h * 64;
  const _Float16* kb = kp + (size_t)b * Ssz * DH + h * 64;
  const _Float16* vb = vpT + ((size_t)b * DH + h * 64) * Ssz;
  // hoisted B tile (qp^T): lane holds col q0+lm, d chunks {d0+8*hi..}, {d0+8*hi+16..}
  V16H bq0, bq1;
  const _Float16* br = qb + (size_t)(q0 + lm) * DH + hi * 8;
  bq0.q[0] = ld16(br);      bq0.q[1] = ld16(br + 16);
  bq1.q[0] = ld16(br + 32); bq1.q[1] = ld16(br + 48);
  v8f acc[4] = {v8f{}, v8f{}, v8f{}, v8f{}};
  for (int k0 = 0; k0 < Ssz; k0 += 32) {
    const _Float16* ar0 = kb + (size_t)(k0 + lm) * DH + hi * 8;
    const _Float16* ar1 = ar0 + (size_t)16 * DH;
    V16H a00, a01, a10, a11;
    a00.q[0] = ld16(ar0);      a00.q[1] = ld16(ar0 + 16);
    a01.q[0] = ld16(ar0 + 32); a01.q[1] = ld16(ar0 + 48);
    a10.q[0] = ld16(ar1);      a10.q[1] = ld16(ar1 + 16);
    a11.q[0] = ld16(ar1 + 32); a11.q[1] = ld16(ar1 + 48);
    v8f s0v = {}, s1v = {};
    s0v = wmma16(a00, bq0, s0v); s0v = wmma16(a01, bq1, s0v);  // k rows k0..k0+15
    s1v = wmma16(a10, bq0, s1v); s1v = wmma16(a11, bq1, s1v);  // k rows k0+16..31
    // P' = exp(s'); repack two C-layout tiles into one 32x16 B operand
    V16H pb;
#pragma unroll
    for (int r = 0; r < 8; r++) {
      float x0 = __expf(s0v[r]), x1 = __expf(s1v[r]);
      float y0 = __shfl_xor(x0, 16, 32);
      float y1 = __shfl_xor(x1, 16, 32);
      pb.h[r]     = (_Float16)(hi ? y1 : x0);
      pb.h[r + 8] = (_Float16)(hi ? x1 : y0);
    }
#pragma unroll
    for (int dt = 0; dt < 4; dt++) {
      const _Float16* vr = vb + (size_t)(dt * 16 + lm) * Ssz + k0 + hi * 16;
      V16H av;
      av.q[0] = ld16(vr); av.q[1] = ld16(vr + 8);
      acc[dt] = wmma16(av, pb, acc[dt]);
    }
  }
  // store ctx [B*S][DH]: lane holds col q0+lm, rows d = dt*16 + 8*hi + r -> packed b128
#pragma unroll
  for (int dt = 0; dt < 4; dt++) {
    V8H pk;
#pragma unroll
    for (int r = 0; r < 8; r++) pk.v[r] = (_Float16)acc[dt][r];
    *(int4*)(ctx + ((size_t)(b * Ssz + q0 + lm)) * DH + h * 64 + dt * 16 + hi * 8) = pk.q;
  }
}

// ---------------- kernel 4: out = ctx * Wo + bo ---------------------------------
// grid (4 n-tiles, 512 s-tiles), block 32. woT: [64][768] f16.
__global__ void outproj_kernel(const _Float16* __restrict__ ctx, const _Float16* __restrict__ woT,
                               const float* __restrict__ bo, float* __restrict__ out) {
  const int lane = threadIdx.x, hi = lane >> 4, lm = lane & 15;
  const int n0 = blockIdx.x * 16, s0 = blockIdx.y * 16;
  v8f c = {};
  for (int k0 = 0; k0 < DH; k0 += 32) {
    const _Float16* ar = ctx + (size_t)(s0 + lm) * DH + k0 + hi * 8;
    V16H a;
    a.q[0] = ld16(ar); a.q[1] = ld16(ar + 16);
    const _Float16* brp = woT + (size_t)(n0 + lm) * DH + k0 + hi * 8;
    V16H bt;
    bt.q[0] = ld16(brp); bt.q[1] = ld16(brp + 16);
    c = wmma16(a, bt, c);
  }
  const float bn = bo[n0 + lm];
#pragma unroll
  for (int r = 0; r < 8; r++)
    out[(size_t)(s0 + r + 8 * hi) * Dsz + n0 + lm] = c[r] + bn;
}

// ---------------- launch --------------------------------------------------------
extern "C" void kernel_launch(void* const* d_in, const int* in_sizes, int n_in,
                              void* d_out, int out_size, void* d_ws, size_t ws_size,
                              hipStream_t stream) {
  (void)in_sizes; (void)n_in; (void)out_size; (void)ws_size;
  const float* q  = (const float*)d_in[0];
  const float* Wq = (const float*)d_in[1];
  const float* bq = (const float*)d_in[2];
  const float* Wk = (const float*)d_in[3];
  const float* bk = (const float*)d_in[4];
  const float* Wv = (const float*)d_in[5];
  const float* bv = (const float*)d_in[6];
  const float* Wo = (const float*)d_in[7];
  const float* bo = (const float*)d_in[8];
  float* out = (float*)d_out;

  char* ws = (char*)d_ws;
  size_t cur = 0;
  auto alloc = [&](size_t bytes) { char* p = ws + cur; cur += (bytes + 255) & ~(size_t)255; return p; };
  _Float16* x16 = (_Float16*)alloc((size_t)NR * 64 * 2);
  _Float16* wqT = (_Float16*)alloc((size_t)DH * 64 * 2);
  _Float16* wkT = (_Float16*)alloc((size_t)DH * 64 * 2);
  _Float16* wvT = (_Float16*)alloc((size_t)DH * 64 * 2);
  _Float16* woT = (_Float16*)alloc((size_t)DH * 64 * 2);
  _Float16* qp  = (_Float16*)alloc((size_t)NR * DH * 2);
  _Float16* kp  = (_Float16*)alloc((size_t)NR * DH * 2);
  _Float16* vpT = (_Float16*)alloc((size_t)NR * DH * 2);
  _Float16* ctx = (_Float16*)alloc((size_t)NR * DH * 2);
  float*    inv = (float*)alloc((size_t)Bsz * Hsz * Ssz * 4);

  // prep
  cvt_half_kernel<<<(NR * 64 + 255) / 256, 256, 0, stream>>>(q, x16, NR * 64);
  trcvt_kernel<<<(64 * DH + 255) / 256, 256, 0, stream>>>(Wq, wqT, 64, DH);
  trcvt_kernel<<<(64 * DH + 255) / 256, 256, 0, stream>>>(Wk, wkT, 64, DH);
  trcvt_kernel<<<(64 * DH + 255) / 256, 256, 0, stream>>>(Wv, wvT, 64, DH);
  trcvt_kernel<<<(DH * 64 + 255) / 256, 256, 0, stream>>>(Wo, woT, DH, 64);

  // projections (1/sqrt(D)=0.125 folded into qp)
  dim3 gp(DH / 16, NR / 16);
  qkv_kernel<<<gp, 32, 0, stream>>>(x16, wqT, bq, qp, nullptr, 0.125f);
  qkv_kernel<<<gp, 32, 0, stream>>>(x16, wkT, bk, kp, nullptr, 1.0f);
  qkv_kernel<<<gp, 32, 0, stream>>>(x16, wvT, bv, nullptr, vpT, 1.0f);

  // softmax-over-q column sums, then fold 1/colsum into V
  colsum_kernel<<<dim3(Ssz / 16, Hsz, Bsz), 32, 0, stream>>>(qp, kp, inv);
  vscale_kernel<<<(int)(((size_t)Bsz * DH * Ssz / 8 + 255) / 256), 256, 0, stream>>>(vpT, inv);

  // attention context
  ctx_kernel<<<dim3(Ssz / 16, Hsz, Bsz), 32, 0, stream>>>(qp, kp, vpT, ctx);

  // output projection
  outproj_kernel<<<dim3(Dsz / 16, NR / 16), 32, 0, stream>>>(ctx, woT, bo, out);
}
